// DeformRoIpoolV2_81853486727769
// MI455X (gfx1250) — matrix operand order repaired and checked by reading, not compiled, not run on hardware
//
#include <hip/hip_runtime.h>
#include <hip/hip_bf16.h>

typedef __attribute__((ext_vector_type(16))) __bf16 v16bf;
typedef __attribute__((ext_vector_type(8)))  float  v8f;

#define B_    8
#define CIN   64
#define COUT  64
#define H_    64
#define W_    64
#define KK    9
#define KDIM  576            // CIN * KK
#define NPIX  4096           // H_ * W_

static __device__ __forceinline__ unsigned short f2bf(float f) {
  unsigned u = __float_as_uint(f);
  u += 0x7FFFu + ((u >> 16) & 1u);   // round-to-nearest-even
  return (unsigned short)(u >> 16);
}

// ---------------- K0: main conv weight fp32 -> bf16 (64 x 576, K = c*9+f order) --
__global__ void wconv_bf16_kernel(const float* __restrict__ w,
                                  unsigned short* __restrict__ wbf) {
  int i = blockIdx.x * blockDim.x + threadIdx.x;
  if (i < COUT * KDIM) wbf[i] = f2bf(w[i]);
}

// ---------------- K1: 27-channel 3x3 conv (18 offset ch + 9 sigmoid-mask ch) -----
__global__ void offmask_kernel(const float* __restrict__ x,
                               const float* __restrict__ w_off, const float* __restrict__ b_off,
                               const float* __restrict__ w_mod, const float* __restrict__ b_mod,
                               float* __restrict__ off_buf, float* __restrict__ mask_buf) {
  int idx = blockIdx.x * blockDim.x + threadIdx.x;
  int pix = idx & (NPIX - 1);
  int t   = idx >> 12;            // b*27 + ch
  int ch  = t % 27;
  int b   = t / 27;
  if (b >= B_) return;
  int y = pix >> 6, xp = pix & 63;

  const float* wsel;
  float acc;
  if (ch < 18) { wsel = w_off + ch * KDIM;        acc = b_off[ch]; }
  else         { wsel = w_mod + (ch - 18) * KDIM; acc = b_mod[ch - 18]; }

  const float* xb = x + (size_t)b * CIN * NPIX;
  for (int c = 0; c < CIN; ++c) {
    const float* xc = xb + c * NPIX;
    const float* wc = wsel + c * KK;
    #pragma unroll
    for (int ky = 0; ky < 3; ++ky) {
      int iy = y + ky - 1;
      if (iy < 0 || iy >= H_) continue;
      #pragma unroll
      for (int kx = 0; kx < 3; ++kx) {
        int ix = xp + kx - 1;
        if (ix < 0 || ix >= W_) continue;
        acc = fmaf(xc[iy * W_ + ix], wc[ky * 3 + kx], acc);
      }
    }
  }
  if (ch < 18) off_buf[((size_t)b * 18 + ch) * NPIX + pix] = acc;
  else         mask_buf[((size_t)b * 9 + (ch - 18)) * NPIX + pix] =
                   1.0f / (1.0f + __expf(-acc));
}

// ---------------- K2: fused bilinear-sample im2col (LDS, bf16) + WMMA GEMM -------
__global__ __launch_bounds__(128)
void deform_wmma_kernel(const float* __restrict__ x,
                        const float* __restrict__ off_buf,
                        const float* __restrict__ mask_buf,
                        const unsigned short* __restrict__ wbf,
                        const float* __restrict__ bias,
                        float* __restrict__ out) {
  __shared__ __align__(16) unsigned short Bt[16][KDIM];   // 18 KB bf16 im2col tile
  __shared__ int   s_iy0[144], s_iy1[144], s_ix0[144], s_ix1[144];
  __shared__ float s_w00[144], s_w01[144], s_w10[144], s_w11[144];

  const int xtile = blockIdx.x;            // 0..3   (16 pixels each)
  const int yrow  = blockIdx.y;            // 0..63
  const int b     = blockIdx.z;            // 0..7
  const int p0    = yrow * W_ + xtile * 16;
  const int tid   = threadIdx.x;

  // ---- stage 1: per-(tap f, pixel n) coords + mask-premultiplied bilinear weights
  for (int e = tid; e < 144; e += 128) {
    int f = e >> 4, n = e & 15;
    int pix = p0 + n;
    float dy = off_buf[((size_t)b * 18 + 2 * f    ) * NPIX + pix];
    float dx = off_buf[((size_t)b * 18 + 2 * f + 1) * NPIX + pix];
    float m  = mask_buf[((size_t)b * 9 + f) * NPIX + pix];
    int ky = f / 3, kx = f - 3 * (f / 3);
    float ys = (float)(yrow - 1 + ky) + dy;
    float xs = (float)(xtile * 16 + n - 1 + kx) + dx;
    float fy0 = floorf(ys), fx0 = floorf(xs);
    int y0 = (int)fy0, x0 = (int)fx0;
    float ay = ys - fy0, ax = xs - fx0;
    bool vy0 = (y0     >= 0) && (y0     < H_);
    bool vy1 = (y0 + 1 >= 0) && (y0 + 1 < H_);
    bool vx0 = (x0     >= 0) && (x0     < W_);
    bool vx1 = (x0 + 1 >= 0) && (x0 + 1 < W_);
    s_w00[e] = (vy0 && vx0) ? (1.f - ay) * (1.f - ax) * m : 0.f;
    s_w01[e] = (vy0 && vx1) ? (1.f - ay) * ax         * m : 0.f;
    s_w10[e] = (vy1 && vx0) ? ay * (1.f - ax)         * m : 0.f;
    s_w11[e] = (vy1 && vx1) ? ay * ax                 * m : 0.f;
    s_iy0[e] = min(max(y0, 0), H_ - 1);
    s_iy1[e] = min(max(y0 + 1, 0), H_ - 1);
    s_ix0[e] = min(max(x0, 0), W_ - 1);
    s_ix1[e] = min(max(x0 + 1, 0), W_ - 1);
  }
  __syncthreads();

  // ---- stage 2: build bf16 im2col tile Bt[n][k], k = c*9 + f
  const float* xb = x + (size_t)b * CIN * NPIX;
  for (int s = tid; s < CIN * 144; s += 128) {   // 72 iters per thread
    int n = s & 15;
    int k = s >> 4;                 // 0..575
    int c = k / 9;
    int f = k - 9 * c;
    int e = f * 16 + n;
    const float* xc = xb + c * NPIX;
    float v = s_w00[e] * xc[s_iy0[e] * W_ + s_ix0[e]]
            + s_w01[e] * xc[s_iy0[e] * W_ + s_ix1[e]]
            + s_w10[e] * xc[s_iy1[e] * W_ + s_ix0[e]]
            + s_w11[e] * xc[s_iy1[e] * W_ + s_ix1[e]];
    Bt[n][k] = f2bf(v);
  }
  __syncthreads();

  // ---- stage 3: 4 waves x (16x16) M-tiles, K=576 in 18 steps of v_wmma bf16
  const int wave = tid >> 5;        // m-tile
  const int lane = tid & 31;
  const int g    = lane >> 4;       // lane group (K half selector)
  const int lm   = lane & 15;       // M row within tile (A) / N column (B)
  const int mrow = wave * 16 + lm;

  v8f acc = {0.f, 0.f, 0.f, 0.f, 0.f, 0.f, 0.f, 0.f};
  union Frag { uint4 u[2]; v16bf v; };

  for (int k0 = 0; k0 < KDIM; k0 += 32) {
    Frag A, Bf;
    // A (16x32 bf16): lane elems 0..7 <-> K = k0 + g*8 + 0..7 ; 8..15 <-> +16
    const unsigned short* ap = wbf + (size_t)mrow * KDIM + k0 + g * 8;
    A.u[0] = *reinterpret_cast<const uint4*>(ap);
    A.u[1] = *reinterpret_cast<const uint4*>(ap + 16);
    // B (32x16 bf16): lane elems 0..7 <-> K = k0 + g*8 + 0..7 ; 8..15 <-> +16, N = lm
    const unsigned short* bp = &Bt[lm][k0 + g * 8];
    Bf.u[0] = *reinterpret_cast<const uint4*>(bp);
    Bf.u[1] = *reinterpret_cast<const uint4*>(bp + 16);
    acc = __builtin_amdgcn_wmma_f32_16x16x32_bf16(
        /*neg_a=*/false, A.v, /*neg_b=*/false, Bf.v,
        /*c_mod=*/(short)0, acc, /*reuse_a=*/false, /*reuse_b=*/false);
  }

  // ---- store: D VGPR r -> M = r + g*8, N = lm
  float* outb = out + (size_t)b * COUT * NPIX;
  #pragma unroll
  for (int r = 0; r < 8; ++r) {
    int o = wave * 16 + r + g * 8;
    outb[(size_t)o * NPIX + p0 + lm] = acc[r] + bias[o];
  }
}

// ---------------------------------------------------------------------------------
extern "C" void kernel_launch(void* const* d_in, const int* in_sizes, int n_in,
                              void* d_out, int out_size, void* d_ws, size_t ws_size,
                              hipStream_t stream) {
  const float* x     = (const float*)d_in[0];
  const float* w_off = (const float*)d_in[1];
  const float* b_off = (const float*)d_in[2];
  const float* w_mod = (const float*)d_in[3];
  const float* b_mod = (const float*)d_in[4];
  const float* w     = (const float*)d_in[5];
  const float* bias  = (const float*)d_in[6];
  float* out = (float*)d_out;

  char* ws = (char*)d_ws;
  float*          off_buf  = (float*)(ws);                        // 8*18*4096 f32
  float*          mask_buf = (float*)(ws + 2359296);              // 8*9*4096  f32
  unsigned short* wbf      = (unsigned short*)(ws + 2359296 + 1179648); // 64*576 bf16

  wconv_bf16_kernel<<<(COUT * KDIM + 255) / 256, 256, 0, stream>>>(w, wbf);
  offmask_kernel<<<(B_ * 27 * NPIX + 255) / 256, 256, 0, stream>>>(
      x, w_off, b_off, w_mod, b_mod, off_buf, mask_buf);

  dim3 grid(4, 64, 8);   // (xtile, row, batch)
  deform_wmma_kernel<<<grid, 128, 0, stream>>>(x, off_buf, mask_buf, wbf, bias, out);
}
